// SelfAttention_58901181497432
// MI455X (gfx1250) — matrix operand order repaired and checked by reading, not compiled, not run on hardware
//
#include <hip/hip_runtime.h>
#include <hip/hip_bf16.h>

// ---------------------------------------------------------------------------
// Self-attention forward for MI455X (gfx1250), wave32 + WMMA bf16.
//   T=2048, B=4, D=1024, H=16, hd=64.
// Pipeline: fp32->bf16 convert | QKV GEMM (WMMA) | flash attention (WMMA +
// online softmax) | output GEMM (WMMA).
// ---------------------------------------------------------------------------

typedef __attribute__((ext_vector_type(16))) __bf16 v16bf;
typedef __attribute__((ext_vector_type(8)))  __bf16 v8bf;
typedef __attribute__((ext_vector_type(8)))  float  v8f;

constexpr int T_  = 2048;
constexpr int B_  = 4;
constexpr int D_  = 1024;
constexpr int H_  = 16;
constexpr int HD_ = 64;
constexpr int TB_ = T_ * B_;   // 8192 rows
constexpr int E_  = 3 * D_;    // 3072 qkv features
constexpr int BH_ = B_ * H_;   // 64 head-batches

// ---- WMMA helpers ----------------------------------------------------------

__device__ __forceinline__ v8f wmma_bf16(v16bf a, v16bf b, v8f c) {
    // D = A(16x32) * B(32x16) + C ; fp32 accumulate
    return __builtin_amdgcn_wmma_f32_16x16x32_bf16(
        /*neg_a=*/false, a, /*neg_b=*/false, b,
        /*c_mod=*/(short)0, c, /*reuse_a=*/false, /*reuse_b=*/false);
}

// A-matrix fragment (16x32 bf16, contiguous K order matching B loads):
// lane(0..15)=row, kh=lane>>4 selects K-subset {kh*16..+15}
__device__ __forceinline__ v16bf load_a_frag(const __bf16* rowp, int kh) {
    return *(const v16bf*)(rowp + kh * 16);
}

// ---- fp32 -> bf16 convert --------------------------------------------------

__global__ void cvt_bf16_kernel(const float* __restrict__ src,
                                __bf16* __restrict__ dst, int n) {
    int idx    = blockIdx.x * blockDim.x + threadIdx.x;
    int stride = gridDim.x * blockDim.x;
    for (int i = idx; i < n; i += stride) dst[i] = (__bf16)src[i];
}

// ---- QKV projection GEMM ---------------------------------------------------
// qkv[m,e] = sum_d x[m,d] * W[e,d] + bias[e] ;  m = t*B+b (row-major [T,B,D]).
// Writes q scaled by hd^-0.5 * log2(e) into [bh][t][hd], k into [bh][t][hd],
// v transposed into [bh][hd][t] (all bf16) so attention B-frags are contiguous.

__global__ void qkv_gemm_kernel(const __bf16* __restrict__ xb,
                                const __bf16* __restrict__ wq,
                                const float*  __restrict__ bias,
                                __bf16* __restrict__ qbuf,
                                __bf16* __restrict__ kbuf,
                                __bf16* __restrict__ vtbuf) {
    const int lane = threadIdx.x & 31;
    const int wave = threadIdx.x >> 5;
    const int lr   = lane & 15;
    const int kh   = lane >> 4;
    const int m0   = blockIdx.x * 32;
    const int n0   = (blockIdx.y * 4 + wave) * 64;

    v8f acc[2][4];
    #pragma unroll
    for (int i = 0; i < 2; ++i)
        #pragma unroll
        for (int j = 0; j < 4; ++j) acc[i][j] = (v8f)0.0f;

    const __bf16* arow0 = xb + (size_t)(m0 + lr) * D_;
    const __bf16* arow1 = xb + (size_t)(m0 + 16 + lr) * D_;
    const __bf16* brow[4];
    #pragma unroll
    for (int nn = 0; nn < 4; ++nn)
        brow[nn] = wq + (size_t)(n0 + nn * 16 + lr) * D_;

    for (int k = 0; k < D_; k += 32) {
        if (k + 128 < D_) {            // global_prefetch hints
            __builtin_prefetch(arow0 + k + 128, 0, 1);
            __builtin_prefetch(brow[0] + k + 128, 0, 1);
        }
        v16bf a0 = load_a_frag(arow0 + k, kh);
        v16bf a1 = load_a_frag(arow1 + k, kh);
        #pragma unroll
        for (int nn = 0; nn < 4; ++nn) {
            v16bf b = *(const v16bf*)(brow[nn] + k + kh * 16);   // 32B/lane
            acc[0][nn] = wmma_bf16(a0, b, acc[0][nn]);
            acc[1][nn] = wmma_bf16(a1, b, acc[1][nn]);
        }
    }

    // epilogue: route into q / k / vT head layouts
    const float qscale = 0.125f * 1.44269504088896340736f;  // hd^-0.5 * log2(e)
    #pragma unroll
    for (int nn = 0; nn < 4; ++nn) {
        int e    = n0 + nn * 16 + lr;
        float bv = bias[e];
        int sec  = e >> 10;            // 0=q 1=k 2=v (uniform per fragment)
        int f    = e & (D_ - 1);
        int h    = f >> 6;
        int hd   = f & 63;
        #pragma unroll
        for (int mg = 0; mg < 2; ++mg) {
            #pragma unroll
            for (int i = 0; i < 8; ++i) {
                int row = m0 + mg * 16 + i + kh * 8;   // C-frag row mapping
                int t   = row >> 2;                    // row = t*B + b, B=4
                int b   = row & 3;
                int bh  = b * H_ + h;
                float va = acc[mg][nn][i] + bv;
                if (sec == 0)
                    qbuf[((size_t)bh * T_ + t) * HD_ + hd] = (__bf16)(va * qscale);
                else if (sec == 1)
                    kbuf[((size_t)bh * T_ + t) * HD_ + hd] = (__bf16)va;
                else
                    vtbuf[((size_t)bh * HD_ + hd) * T_ + t] = (__bf16)va;
            }
        }
    }
}

// ---- Flash attention -------------------------------------------------------
// One wave per 16-query tile of one head-batch. Streams 32-key blocks:
//   S(16x32) = Q(16x64) K^T         (4 WMMAs)
//   online softmax in fp32 regs (exp2-based; scale pre-folded into Q)
//   O += P(16x32) V(32x64)          (4 WMMAs, P transposed via LDS)

__global__ void attn_kernel(const __bf16* __restrict__ qbuf,
                            const __bf16* __restrict__ kbuf,
                            const __bf16* __restrict__ vtbuf,
                            __bf16* __restrict__ abuf) {
    __shared__ __align__(128) __bf16 Pl[4][16][32];   // 1KB per wave

    const int lane = threadIdx.x & 31;
    const int wave = threadIdx.x >> 5;
    const int lr   = lane & 15;
    const int kh   = lane >> 4;
    const int bh   = blockIdx.x;
    const int q0   = (blockIdx.y * 4 + wave) * 16;

    const __bf16* qrow  = qbuf + ((size_t)bh * T_ + (q0 + lr)) * HD_;
    const __bf16* kbase = kbuf + (size_t)bh * T_ * HD_;
    const __bf16* vbase = vtbuf + (size_t)bh * HD_ * T_;

    v16bf Aq0 = load_a_frag(qrow, kh);        // d = 0..31
    v16bf Aq1 = load_a_frag(qrow + 32, kh);   // d = 32..63

    v8f O[4];
    #pragma unroll
    for (int nn = 0; nn < 4; ++nn) O[nn] = (v8f)0.0f;
    float mrow[8], lsum[8];
    #pragma unroll
    for (int i = 0; i < 8; ++i) { mrow[i] = -1e30f; lsum[i] = 0.0f; }

    for (int j0 = 0; j0 < T_; j0 += 32) {
        // --- scores S = Q K^T over this 32-key block -----------------------
        v8f S[2];
        #pragma unroll
        for (int g = 0; g < 2; ++g) {
            const __bf16* krow = kbase + (size_t)(j0 + g * 16 + lr) * HD_;
            v16bf b0 = *(const v16bf*)(krow + kh * 16);        // d 0..31
            v16bf b1 = *(const v16bf*)(krow + 32 + kh * 16);   // d 32..63
            v8f s = (v8f)0.0f;
            s    = wmma_bf16(Aq0, b0, s);
            S[g] = wmma_bf16(Aq1, b1, s);
        }
        // --- online softmax (per-row butterflies inside 16-lane halves) ----
        #pragma unroll
        for (int i = 0; i < 8; ++i) {
            float s0 = S[0][i], s1 = S[1][i];
            float cm = fmaxf(s0, s1);
            cm = fmaxf(cm, __shfl_xor(cm, 1, 32));
            cm = fmaxf(cm, __shfl_xor(cm, 2, 32));
            cm = fmaxf(cm, __shfl_xor(cm, 4, 32));
            cm = fmaxf(cm, __shfl_xor(cm, 8, 32));
            float mnew = fmaxf(mrow[i], cm);
            float sc   = exp2f(mrow[i] - mnew);
            float p0   = exp2f(s0 - mnew);
            float p1   = exp2f(s1 - mnew);
            float ps   = p0 + p1;
            ps += __shfl_xor(ps, 1, 32);
            ps += __shfl_xor(ps, 2, 32);
            ps += __shfl_xor(ps, 4, 32);
            ps += __shfl_xor(ps, 8, 32);
            lsum[i] = lsum[i] * sc + ps;
            mrow[i] = mnew;
            O[0][i] *= sc; O[1][i] *= sc; O[2][i] *= sc; O[3][i] *= sc;
            int r = i + kh * 8;                 // C-frag (row, col) -> LDS
            Pl[wave][r][lr]      = (__bf16)p0;
            Pl[wave][r][16 + lr] = (__bf16)p1;
        }
        __builtin_amdgcn_wave_barrier();
        asm volatile("s_wait_dscnt 0" ::: "memory");   // LDS stores visible
        v16bf Ap = load_a_frag(&Pl[wave][lr][0], kh);  // P in A-frag layout
        asm volatile("" ::: "memory");                 // keep read before next store
        // --- O += P * V ----------------------------------------------------
        #pragma unroll
        for (int nn = 0; nn < 4; ++nn) {
            const __bf16* vrow =
                vbase + (size_t)(nn * 16 + lr) * T_ + j0 + kh * 16;  // 32B/lane
            v16bf bv = *(const v16bf*)vrow;
            O[nn] = wmma_bf16(Ap, bv, O[nn]);
        }
    }

    // --- normalize and write back into [t, b, d] row-major bf16 ------------
    const int b = bh >> 4, h = bh & 15;
    #pragma unroll
    for (int i = 0; i < 8; ++i) {
        float rl = 1.0f / lsum[i];
        int t = q0 + i + kh * 8;
        #pragma unroll
        for (int nn = 0; nn < 4; ++nn) {
            int d = h * 64 + nn * 16 + lr;
            abuf[((size_t)t * B_ + b) * D_ + d] = (__bf16)(O[nn][i] * rl);
        }
    }
}

// ---- Output projection GEMM ------------------------------------------------
// out[m,e] = sum_d attn[m,d] * out_w[e,d] + out_b[e]   (fp32 out)

__global__ void out_gemm_kernel(const __bf16* __restrict__ abuf,
                                const __bf16* __restrict__ wob,
                                const float*  __restrict__ outb,
                                float* __restrict__ out) {
    const int lane = threadIdx.x & 31;
    const int wave = threadIdx.x >> 5;
    const int lr   = lane & 15;
    const int kh   = lane >> 4;
    const int m0   = blockIdx.x * 32;
    const int n0   = (blockIdx.y * 4 + wave) * 64;

    v8f acc[2][4];
    #pragma unroll
    for (int i = 0; i < 2; ++i)
        #pragma unroll
        for (int j = 0; j < 4; ++j) acc[i][j] = (v8f)0.0f;

    const __bf16* arow0 = abuf + (size_t)(m0 + lr) * D_;
    const __bf16* arow1 = abuf + (size_t)(m0 + 16 + lr) * D_;
    const __bf16* brow[4];
    #pragma unroll
    for (int nn = 0; nn < 4; ++nn)
        brow[nn] = wob + (size_t)(n0 + nn * 16 + lr) * D_;

    for (int k = 0; k < D_; k += 32) {
        if (k + 128 < D_) {
            __builtin_prefetch(arow0 + k + 128, 0, 1);
            __builtin_prefetch(brow[0] + k + 128, 0, 1);
        }
        v16bf a0 = load_a_frag(arow0 + k, kh);
        v16bf a1 = load_a_frag(arow1 + k, kh);
        #pragma unroll
        for (int nn = 0; nn < 4; ++nn) {
            v16bf b = *(const v16bf*)(brow[nn] + k + kh * 16);
            acc[0][nn] = wmma_bf16(a0, b, acc[0][nn]);
            acc[1][nn] = wmma_bf16(a1, b, acc[1][nn]);
        }
    }

    #pragma unroll
    for (int nn = 0; nn < 4; ++nn) {
        int e    = n0 + nn * 16 + lr;
        float bv = outb[e];
        #pragma unroll
        for (int mg = 0; mg < 2; ++mg) {
            #pragma unroll
            for (int i = 0; i < 8; ++i) {
                int row = m0 + mg * 16 + i + kh * 8;
                out[(size_t)row * D_ + e] = acc[mg][nn][i] + bv;
            }
        }
    }
}

// ---- host-side launcher ----------------------------------------------------

extern "C" void kernel_launch(void* const* d_in, const int* in_sizes, int n_in,
                              void* d_out, int out_size, void* d_ws, size_t ws_size,
                              hipStream_t stream) {
    const float* x   = (const float*)d_in[0];   // [T,B,D]
    const float* ipw = (const float*)d_in[1];   // [3D,D]
    const float* ipb = (const float*)d_in[2];   // [3D]
    const float* ow  = (const float*)d_in[3];   // [D,D]
    const float* ob  = (const float*)d_in[4];   // [D]
    float* out = (float*)d_out;                 // [T,B,D]

    // workspace carve-up (bf16 elements), ~88 MB total
    __bf16* ws    = (__bf16*)d_ws;
    __bf16* xb    = ws;                                   // [8192][1024]
    __bf16* wqkv  = xb    + (size_t)TB_ * D_;             // [3072][1024]
    __bf16* wout  = wqkv  + (size_t)E_ * D_;              // [1024][1024]
    __bf16* qbuf  = wout  + (size_t)D_ * D_;              // [64][2048][64]
    __bf16* kbuf  = qbuf  + (size_t)BH_ * T_ * HD_;       // [64][2048][64]
    __bf16* vtbuf = kbuf  + (size_t)BH_ * T_ * HD_;       // [64][64][2048]
    __bf16* abuf  = vtbuf + (size_t)BH_ * HD_ * T_;       // [8192][1024]

    cvt_bf16_kernel<<<1024, 256, 0, stream>>>(x, xb, TB_ * D_);
    cvt_bf16_kernel<<<1024, 256, 0, stream>>>(ipw, wqkv, E_ * D_);
    cvt_bf16_kernel<<<512,  256, 0, stream>>>(ow, wout, D_ * D_);

    // M tiles: 8192/32 = 256 ; N wave-tiles of 64 -> 3072/64/4 = 12 blocks
    qkv_gemm_kernel<<<dim3(256, 12), 128, 0, stream>>>(xb, wqkv, ipb,
                                                       qbuf, kbuf, vtbuf);
    // 64 head-batches x (2048/16 q-tiles / 4 waves) = (64, 32)
    attn_kernel<<<dim3(64, 32), 128, 0, stream>>>(qbuf, kbuf, vtbuf, abuf);
    // 1024/64/4 = 4 N-blocks
    out_gemm_kernel<<<dim3(256, 4), 128, 0, stream>>>(abuf, wout, ob, out);
}
